// SuperChargedSeqTransformer_88158498718434
// MI455X (gfx1250) — matrix operand order repaired and checked by reading, not compiled, not run on hardware
//
#include <hip/hip_runtime.h>

// ---------------------------------------------------------------------------
// CDNA5 (gfx1250) WMMA-based forward pass for SuperChargedSeqTransformer.
// wave32; matrix ops via __builtin_amdgcn_wmma_f32_16x16x32_f16.
// ---------------------------------------------------------------------------

typedef _Float16 v16h __attribute__((ext_vector_type(16)));
typedef float    v8f  __attribute__((ext_vector_type(8)));

#define ACT_NONE 0
#define ACT_SILU 1
#define ACT_SIGM 2

// epilogue helper: bias -> activation -> residual -> bounds-checked store
__device__ __forceinline__
void epi_store(v8f acc, int mb, int nb, int r, int hi,
               float* __restrict__ C, int ldc,
               const float* __restrict__ bias, const float* __restrict__ res,
               int M, int N, int act)
{
    for (int e = 0; e < 8; ++e) {
        int m = mb + e + hi * 8;       // C/D layout: VGPR e -> M = e (+8 for hi half)
        int n = nb + r;
        if (m < M && n < N) {
            float v = acc[e];
            if (bias) v += bias[n];
            if (act == ACT_SILU)      v = v / (1.0f + __expf(-v));
            else if (act == ACT_SIGM) v = 1.0f / (1.0f + __expf(-v));
            if (res) v += res[(size_t)m * ldc + n];
            C[(size_t)m * ldc + n] = v;
        }
    }
}

// ---------------------------------------------------------------------------
// Generic tiled GEMM:  C[M,N] = epilogue(A[M,K] @ W[K,N])
//  - block = 256 threads = 8 waves; block tile 64(M) x 128(N); K step 32
//  - each wave owns a 32x32 tile => 4 x v_wmma_f32_16x16x32_f16 per K step
//  - fast path: float4 (b128) staging when the tile is fully in range
// ---------------------------------------------------------------------------
__global__ __launch_bounds__(256)
void gemm_wmma_f16(const float* __restrict__ A, int lda,
                   const float* __restrict__ W, int ldw,
                   float* __restrict__ C, int ldc,
                   const float* __restrict__ bias,
                   const float* __restrict__ res,
                   int M, int N, int K, int act)
{
    __shared__ _Float16 As[64][40];    // [m][k], padded
    __shared__ _Float16 Ws[128][40];   // [n][k] (transposed), padded

    const int tid  = threadIdx.x;
    const int wave = tid >> 5;
    const int lane = tid & 31;
    const int wm = wave >> 2;          // 0..1
    const int wn = wave & 3;           // 0..3
    const int m0 = blockIdx.y * 64;
    const int n0 = blockIdx.x * 128;
    const int r  = lane & 15;
    const int hi = lane >> 4;

    const bool fullM = (m0 + 64 <= M);
    const bool fullN = (n0 + 128 <= N);

    v8f acc00 = {}, acc01 = {}, acc10 = {}, acc11 = {};

    for (int k0 = 0; k0 < K; k0 += 32) {
        const bool fullK = (k0 + 32 <= K);

        // ---- stage A tile (64 x 32) ----
        if (fullM && fullK) {
            int m  = tid >> 2;               // 0..63
            int kk = (tid & 3) * 8;          // 0,8,16,24
            const float4* src = (const float4*)&A[(size_t)(m0 + m) * lda + k0 + kk];
            float4 v0 = src[0], v1 = src[1];
            As[m][kk + 0] = (_Float16)v0.x; As[m][kk + 1] = (_Float16)v0.y;
            As[m][kk + 2] = (_Float16)v0.z; As[m][kk + 3] = (_Float16)v0.w;
            As[m][kk + 4] = (_Float16)v1.x; As[m][kk + 5] = (_Float16)v1.y;
            As[m][kk + 6] = (_Float16)v1.z; As[m][kk + 7] = (_Float16)v1.w;
            if (k0 + 32 < K) __builtin_prefetch(&A[(size_t)(m0 + m) * lda + k0 + 32 + kk], 0, 0);
        } else {
            for (int i = tid; i < 64 * 32; i += 256) {
                int m = i >> 5, k = i & 31;
                float v = 0.0f;
                if (m0 + m < M && k0 + k < K) v = A[(size_t)(m0 + m) * lda + (k0 + k)];
                As[m][k] = (_Float16)v;
            }
        }

        // ---- stage W tile transposed (128 cols x 32 k) ----
        if (fullN && fullK) {
            int k  = tid >> 3;               // 0..31
            int nn = (tid & 7) * 16;         // 0..112
            const float4* src = (const float4*)&W[(size_t)(k0 + k) * ldw + n0 + nn];
            for (int q4 = 0; q4 < 4; ++q4) {
                float4 v = src[q4];
                Ws[nn + q4 * 4 + 0][k] = (_Float16)v.x;
                Ws[nn + q4 * 4 + 1][k] = (_Float16)v.y;
                Ws[nn + q4 * 4 + 2][k] = (_Float16)v.z;
                Ws[nn + q4 * 4 + 3][k] = (_Float16)v.w;
            }
            if (k0 + 32 < K) __builtin_prefetch(&W[(size_t)(k0 + 32 + k) * ldw + n0 + nn], 0, 0);
        } else {
            for (int i = tid; i < 128 * 32; i += 256) {
                int n = i & 127, k = i >> 7;
                float v = 0.0f;
                if (k0 + k < K && n0 + n < N) v = W[(size_t)(k0 + k) * ldw + (n0 + n)];
                Ws[n][k] = (_Float16)v;
            }
        }
        __syncthreads();

        // ---- build operands with contiguous k = 16*hi + e (A and B alike) ----
        v16h a0, a1, b0, b1;
        for (int e = 0; e < 16; ++e) {
            int ka = hi * 16 + e;                               // A: lane-half K split
            a0[e] = As[wm * 32 + r][ka];
            a1[e] = As[wm * 32 + 16 + r][ka];
            int kb = hi * 16 + e;                               // B: column-per-lane
            b0[e] = Ws[wn * 32 + r][kb];
            b1[e] = Ws[wn * 32 + 16 + r][kb];
        }
        acc00 = __builtin_amdgcn_wmma_f32_16x16x32_f16(false, a0, false, b0, (short)0, acc00, false, false);
        acc01 = __builtin_amdgcn_wmma_f32_16x16x32_f16(false, a0, false, b1, (short)0, acc01, false, false);
        acc10 = __builtin_amdgcn_wmma_f32_16x16x32_f16(false, a1, false, b0, (short)0, acc10, false, false);
        acc11 = __builtin_amdgcn_wmma_f32_16x16x32_f16(false, a1, false, b1, (short)0, acc11, false, false);
        __syncthreads();
    }

    epi_store(acc00, m0 + wm * 32,      n0 + wn * 32,      r, hi, C, ldc, bias, res, M, N, act);
    epi_store(acc01, m0 + wm * 32,      n0 + wn * 32 + 16, r, hi, C, ldc, bias, res, M, N, act);
    epi_store(acc10, m0 + wm * 32 + 16, n0 + wn * 32,      r, hi, C, ldc, bias, res, M, N, act);
    epi_store(acc11, m0 + wm * 32 + 16, n0 + wn * 32 + 16, r, hi, C, ldc, bias, res, M, N, act);
}

// ---------------------------------------------------------------------------
// Fused attention for one (batch, head): WMMA QK^T -> LDS softmax -> WMMA P*V.
// grid = (B*NH, T/16), block = 128 (4 waves). T=512, DH=32.
// K staged to LDS 64 keys at a time; V staged transposed per 32-key step.
// ---------------------------------------------------------------------------
__global__ __launch_bounds__(128)
void attn_wmma(const float* __restrict__ qp, int qrs,
               const float* __restrict__ kp, int krs,
               const float* __restrict__ vp, int vrs,
               float* __restrict__ op,
               int kvdiv, int causal,
               const float* __restrict__ temp_ptr, float scale_const)
{
    const int T = 512, DH = 32, OD = 512;
    __shared__ _Float16 qs[16][40];
    __shared__ _Float16 ks[64][40];     // staged keys  [key][dh]
    __shared__ _Float16 vst[32][40];    // staged values, transposed [dh][key]
    __shared__ float    ss[16][516];
    __shared__ _Float16 ps[16][520];
    __shared__ float    red[16][8];
    __shared__ float    red2[16][8];

    const int bh = blockIdx.x;
    const int qt = blockIdx.y;
    const int b  = bh >> 4;
    const int h  = bh & 15;
    const int kvh = h / kvdiv;

    const int tid  = threadIdx.x;
    const int wave = tid >> 5;
    const int lane = tid & 31;
    const int r  = lane & 15;
    const int hi = lane >> 4;

    float scale = scale_const;
    if (temp_ptr) scale = fminf(__expf(*temp_ptr), 50.0f);

    // ---- stage Q tile (16 x 32) as f16: one float4 per thread ----
    {
        int m = tid >> 3, d4 = (tid & 7) * 4;
        float4 v = *(const float4*)&qp[(size_t)(b * T + qt * 16 + m) * qrs + h * DH + d4];
        qs[m][d4 + 0] = (_Float16)v.x; qs[m][d4 + 1] = (_Float16)v.y;
        qs[m][d4 + 2] = (_Float16)v.z; qs[m][d4 + 3] = (_Float16)v.w;
    }
    __syncthreads();

    v16h a;
    for (int e = 0; e < 16; ++e) {
        int ka = hi * 16 + e;
        a[e] = qs[r][ka];
    }

    // ---- scores: groups of 64 keys staged to LDS, one 16-key tile per wave ----
    for (int ktg = 0; ktg < 8; ++ktg) {
        {
            int kl = tid >> 1;                    // 0..63 local key
            int d0 = (tid & 1) * 16;              // dh segment
            const float4* src = (const float4*)&kp[(size_t)(b * T + ktg * 64 + kl) * krs
                                                   + kvh * DH + d0];
            for (int q4 = 0; q4 < 4; ++q4) {
                float4 v = src[q4];
                ks[kl][d0 + q4 * 4 + 0] = (_Float16)v.x;
                ks[kl][d0 + q4 * 4 + 1] = (_Float16)v.y;
                ks[kl][d0 + q4 * 4 + 2] = (_Float16)v.z;
                ks[kl][d0 + q4 * 4 + 3] = (_Float16)v.w;
            }
        }
        __syncthreads();
        {
            v16h bk;
            for (int e = 0; e < 16; ++e) {
                int kk = hi * 16 + e;
                bk[e] = ks[wave * 16 + r][kk];
            }
            v8f sacc = {};
            sacc = __builtin_amdgcn_wmma_f32_16x16x32_f16(false, a, false, bk,
                                                          (short)0, sacc, false, false);
            int kt = ktg * 4 + wave;
            for (int e = 0; e < 8; ++e) {
                int m = e + hi * 8;
                int gq = qt * 16 + m, gk = kt * 16 + r;
                float v = sacc[e] * scale;
                if (causal && gk > gq) v = -1e30f;
                ss[m][kt * 16 + r] = v;
            }
        }
        __syncthreads();
    }

    // ---- row softmax in LDS (16 rows, 8 thread-groups of 64 cols) ----
    const int row = tid & 15, grp = tid >> 4;
    float mx = -1e30f;
    for (int c = grp * 64; c < grp * 64 + 64; ++c) mx = fmaxf(mx, ss[row][c]);
    red[row][grp] = mx;
    __syncthreads();
    float m = -1e30f;
    for (int g = 0; g < 8; ++g) m = fmaxf(m, red[row][g]);
    float psum = 0.0f;
    for (int c = grp * 64; c < grp * 64 + 64; ++c) {
        float e = __expf(ss[row][c] - m);
        ss[row][c] = e;
        psum += e;
    }
    red2[row][grp] = psum;
    __syncthreads();
    float tot = 0.0f;
    for (int g = 0; g < 8; ++g) tot += red2[row][g];
    float inv = 1.0f / tot;
    for (int c = grp * 64; c < grp * 64 + 64; ++c)
        ps[row][c] = (_Float16)(ss[row][c] * inv);
    __syncthreads();

    // ---- P @ V : stage V tile transposed, waves 0,1 own the two DH tiles ----
    const int n16 = wave;    // valid for waves 0,1
    v8f oacc = {};
    for (int kt2 = 0; kt2 < T / 32; ++kt2) {
        {
            int kl = tid >> 2;                    // 0..31 local key
            int d0 = (tid & 3) * 8;               // 8 dh per thread
            const float4* src = (const float4*)&vp[(size_t)(b * T + kt2 * 32 + kl) * vrs
                                                   + kvh * DH + d0];
            float4 v0 = src[0], v1 = src[1];
            vst[d0 + 0][kl] = (_Float16)v0.x; vst[d0 + 1][kl] = (_Float16)v0.y;
            vst[d0 + 2][kl] = (_Float16)v0.z; vst[d0 + 3][kl] = (_Float16)v0.w;
            vst[d0 + 4][kl] = (_Float16)v1.x; vst[d0 + 5][kl] = (_Float16)v1.y;
            vst[d0 + 6][kl] = (_Float16)v1.z; vst[d0 + 7][kl] = (_Float16)v1.w;
        }
        __syncthreads();
        if (wave < 2) {
            v16h pa, vb;
            for (int e = 0; e < 16; ++e) {
                int ka = hi * 16 + e;
                pa[e] = ps[r][kt2 * 32 + ka];
                vb[e] = vst[n16 * 16 + r][hi * 16 + e];
            }
            oacc = __builtin_amdgcn_wmma_f32_16x16x32_f16(false, pa, false, vb,
                                                          (short)0, oacc, false, false);
        }
        __syncthreads();
    }
    if (wave < 2) {
        for (int e = 0; e < 8; ++e) {
            int mq = e + hi * 8;
            op[(size_t)(b * T + qt * 16 + mq) * OD + h * DH + n16 * 16 + r] = oacc[e];
        }
    }
}

// ---------------------------------------------------------------------------
// LayerNorm over D=512, one block (128 threads) per row; in-place safe.
// ---------------------------------------------------------------------------
__global__ __launch_bounds__(128)
void layernorm_k(const float* __restrict__ x, float* __restrict__ y,
                 const float* __restrict__ w, const float* __restrict__ b, int D)
{
    const int row = blockIdx.x;
    const float* xr = x + (size_t)row * D;
    float* yr = y + (size_t)row * D;
    __shared__ float s1[128], s2[128];
    float a = 0.0f, q = 0.0f;
    for (int d = threadIdx.x; d < D; d += 128) { float v = xr[d]; a += v; q += v * v; }
    s1[threadIdx.x] = a; s2[threadIdx.x] = q;
    __syncthreads();
    for (int off = 64; off > 0; off >>= 1) {
        if (threadIdx.x < off) { s1[threadIdx.x] += s1[threadIdx.x + off];
                                 s2[threadIdx.x] += s2[threadIdx.x + off]; }
        __syncthreads();
    }
    float mean = s1[0] / D;
    float var  = s2[0] / D - mean * mean;
    float inv  = rsqrtf(var + 1e-5f);
    for (int d = threadIdx.x; d < D; d += 128) {
        float v = (xr[d] - mean) * inv * w[d];
        if (b) v += b[d];
        yr[d] = v;
    }
}

// per-head qk-norm over DH=32 (weight only, no bias), in place
__global__ void qknorm_k(float* __restrict__ x, const float* __restrict__ w,
                         int rows, int heads, int rowstride)
{
    int idx = blockIdx.x * blockDim.x + threadIdx.x;
    if (idx >= rows * heads) return;
    int row = idx / heads, h = idx % heads;
    float* p = x + (size_t)row * rowstride + h * 32;
    float a = 0.0f, q = 0.0f;
    for (int d = 0; d < 32; ++d) { float v = p[d]; a += v; q += v * v; }
    float mean = a / 32.0f;
    float var  = q / 32.0f - mean * mean;
    float inv  = rsqrtf(var + 1e-5f);
    for (int d = 0; d < 32; ++d) p[d] = (p[d] - mean) * inv * w[d];
}

// patch-mean token embedding + pe_proj (K=16) + positional embedding
__global__ __launch_bounds__(128)
void embed_k(const int* __restrict__ x, const float* __restrict__ tok_emb,
             const float* __restrict__ pw, const float* __restrict__ pb,
             const float* __restrict__ pos, float* __restrict__ z)
{
    const int bt = blockIdx.x;       // b*512 + t
    const int b = bt >> 9, t = bt & 511;
    __shared__ float feat[16];
    if (threadIdx.x < 16) {
        float s = 0.0f;
        for (int i = 0; i < 16; ++i) {
            int tok = x[b * 8192 + t * 16 + i];
            s += tok_emb[tok * 16 + threadIdx.x];
        }
        feat[threadIdx.x] = s * (1.0f / 16.0f);
    }
    __syncthreads();
    for (int d = threadIdx.x; d < 512; d += 128) {
        float v = pb[d] + pos[t * 512 + d];
        for (int c = 0; c < 16; ++c) v += feat[c] * pw[c * 512 + d];
        z[(size_t)bt * 512 + d] = v;
    }
}

// mem attention scores: sc[b,m,t] = mem[m].z[b,t] / sqrt(512)
__global__ __launch_bounds__(128)
void mem_score_k(const float* __restrict__ mem, const float* __restrict__ z,
                 float* __restrict__ sc)
{
    const int bm = blockIdx.x;       // b*20 + m
    const int b = bm / 20, mrow = bm % 20;
    const float* mv = mem + mrow * 512;
    for (int t = threadIdx.x; t < 512; t += 128) {
        const float* zr = z + (size_t)(b * 512 + t) * 512;
        float s = 0.0f;
        for (int d = 0; d < 512; ++d) s += mv[d] * zr[d];
        sc[(size_t)bm * 512 + t] = s * 0.04419417382f;   // 1/sqrt(512)
    }
}

// softmax over t + ctx[b,m,:] = sum_t aw * z[b,t,:]
__global__ __launch_bounds__(128)
void mem_ctx_k(const float* __restrict__ z, float* __restrict__ sc,
               float* __restrict__ ctx)
{
    const int bm = blockIdx.x;
    const int b = bm / 20;
    float* row = sc + (size_t)bm * 512;
    __shared__ float s1[128];
    float mx = -1e30f;
    for (int t = threadIdx.x; t < 512; t += 128) mx = fmaxf(mx, row[t]);
    s1[threadIdx.x] = mx; __syncthreads();
    for (int off = 64; off > 0; off >>= 1) {
        if (threadIdx.x < off) s1[threadIdx.x] = fmaxf(s1[threadIdx.x], s1[threadIdx.x + off]);
        __syncthreads();
    }
    float m = s1[0]; __syncthreads();
    float ps = 0.0f;
    for (int t = threadIdx.x; t < 512; t += 128) { float e = __expf(row[t] - m); row[t] = e; ps += e; }
    s1[threadIdx.x] = ps; __syncthreads();
    for (int off = 64; off > 0; off >>= 1) {
        if (threadIdx.x < off) s1[threadIdx.x] += s1[threadIdx.x + off];
        __syncthreads();
    }
    float inv = 1.0f / s1[0];
    for (int d = threadIdx.x; d < 512; d += 128) {
        float acc = 0.0f;
        for (int t = 0; t < 512; ++t) acc += row[t] * z[(size_t)(b * 512 + t) * 512 + d];
        ctx[(size_t)bm * 512 + d] = acc * inv;
    }
}

__global__ void mem_mean_k(const float* __restrict__ ctx, float* __restrict__ cmean)
{
    int idx = blockIdx.x * blockDim.x + threadIdx.x;
    if (idx >= 4 * 512) return;
    int b = idx >> 9, d = idx & 511;
    float s = 0.0f;
    for (int mrow = 0; mrow < 20; ++mrow) s += ctx[(size_t)(b * 20 + mrow) * 512 + d];
    cmean[idx] = s * (1.0f / 20.0f);
}

__global__ void xcat_k(const float* __restrict__ z, const float* __restrict__ mi,
                       float* __restrict__ xcat)
{
    size_t idx = (size_t)blockIdx.x * blockDim.x + threadIdx.x;
    if (idx >= (size_t)2048 * 1024) return;
    int rowi = (int)(idx >> 10), d2 = (int)(idx & 1023);
    int b = rowi >> 9;
    xcat[idx] = (d2 < 512) ? z[(size_t)rowi * 512 + d2] : mi[b * 512 + d2 - 512];
}

__global__ void gateupd_k(float* __restrict__ z, const float* __restrict__ gate,
                          const float* __restrict__ mi)
{
    size_t idx = (size_t)blockIdx.x * blockDim.x + threadIdx.x;
    if (idx >= (size_t)2048 * 512) return;
    int rowi = (int)(idx >> 9), d = (int)(idx & 511), b = rowi >> 9;
    z[idx] += gate[idx] * mi[b * 512 + d];
}

// |rfft(z, axis=t)|[:, :32, :64]   (naive DFT, tiny)
__global__ void dft_k(const float* __restrict__ z, float* __restrict__ feats)
{
    int idx = blockIdx.x * blockDim.x + threadIdx.x;
    if (idx >= 4 * 32 * 64) return;
    int b = idx / (32 * 64), rest = idx % (32 * 64), f = rest / 64, d = rest % 64;
    float re = 0.0f, im = 0.0f;
    float w0 = -2.0f * 3.14159265358979f * (float)f / 512.0f;
    for (int t = 0; t < 512; ++t) {
        float v = z[(size_t)(b * 512 + t) * 512 + d];
        float s, c;
        __sincosf(w0 * (float)t, &s, &c);
        re += v * c;
        im += v * s;
    }
    feats[idx] = sqrtf(re * re + im * im);
}

__global__ void fmean_k(const float* __restrict__ feats, float* __restrict__ mfeat)
{
    int idx = blockIdx.x * blockDim.x + threadIdx.x;
    if (idx >= 4 * 64) return;
    int b = idx >> 6, d = idx & 63;
    float s = 0.0f;
    for (int f = 0; f < 32; ++f) s += feats[(b * 32 + f) * 64 + d];
    mfeat[idx] = s * (1.0f / 32.0f);
}

__global__ void addbc_k(float* __restrict__ z, const float* __restrict__ fo)
{
    size_t idx = (size_t)blockIdx.x * blockDim.x + threadIdx.x;
    if (idx >= (size_t)2048 * 512) return;
    int rowi = (int)(idx >> 9), d = (int)(idx & 511), b = rowi >> 9;
    z[idx] += fo[b * 512 + d];
}

// ---------------------------------------------------------------------------
// Host orchestration
// ---------------------------------------------------------------------------
static inline void launch_gemm(hipStream_t st,
                               const float* A, int lda, const float* W, int ldw,
                               float* C, int ldc, const float* bias, const float* res,
                               int M, int N, int K, int act)
{
    dim3 grid((N + 127) / 128, (M + 63) / 64);
    gemm_wmma_f16<<<grid, 256, 0, st>>>(A, lda, W, ldw, C, ldc, bias, res, M, N, K, act);
}

extern "C" void kernel_launch(void* const* d_in, const int* in_sizes, int n_in,
                              void* d_out, int out_size, void* d_ws, size_t ws_size,
                              hipStream_t stream)
{
    (void)in_sizes; (void)n_in; (void)out_size; (void)ws_size;

    const int*   x       = (const int*)d_in[0];
    const float* tok_emb = (const float*)d_in[1];
    const float* pe_w    = (const float*)d_in[2];
    const float* pe_b    = (const float*)d_in[3];
    const float* pos     = (const float*)d_in[4];
    const float* memtok  = (const float*)d_in[5];
    const float* mem_pw  = (const float*)d_in[6];
    const float* mem_pb  = (const float*)d_in[7];
    const float* gate_w  = (const float*)d_in[8];
    const float* gate_b  = (const float*)d_in[9];
    const float* hn_w    = (const float*)d_in[10];
    const float* hn_b    = (const float*)d_in[11];
    const float* h_inw   = (const float*)d_in[12];
    const float* h_inb   = (const float*)d_in[13];
    const float* h_outw  = (const float*)d_in[14];
    const float* h_outb  = (const float*)d_in[15];
    const float* g_n1w   = (const float*)d_in[16];
    const float* g_n1b   = (const float*)d_in[17];
    const float* g_n2w   = (const float*)d_in[18];
    const float* g_n2b   = (const float*)d_in[19];
    const float* g_qw    = (const float*)d_in[20];
    const float* g_kw    = (const float*)d_in[21];
    const float* g_vw    = (const float*)d_in[22];
    const float* g_ow    = (const float*)d_in[23];
    const float* g_qnw   = (const float*)d_in[24];
    const float* g_knw   = (const float*)d_in[25];
    const float* g_temp  = (const float*)d_in[26];
    const float* g_lnw   = (const float*)d_in[27];
    const float* g_lnb   = (const float*)d_in[28];
    const float* g_w1    = (const float*)d_in[29];
    const float* g_b1    = (const float*)d_in[30];
    const float* g_w2    = (const float*)d_in[31];
    const float* g_b2    = (const float*)d_in[32];
    const float* f_fw    = (const float*)d_in[33];
    const float* f_fb    = (const float*)d_in[34];
    const float* f_ow    = (const float*)d_in[35];
    const float* f_ob    = (const float*)d_in[36];
    const float* lnf_w   = (const float*)d_in[37];
    const float* lnf_b   = (const float*)d_in[38];
    const float* hd_w1   = (const float*)d_in[39];
    const float* hd_b1   = (const float*)d_in[40];
    const float* hd_w2   = (const float*)d_in[41];
    const float* hd_b2   = (const float*)d_in[42];
    const float* hd_w3   = (const float*)d_in[43];
    const float* hd_b3   = (const float*)d_in[44];

    const int M = 2048;            // B*T rows

    // ---- workspace bump allocator (floats) ----
    float* p = (float*)d_ws;
    float* z    = p; p += (size_t)M * 512;
    float* h    = p; p += (size_t)M * 512;
    float* qb   = p; p += (size_t)M * 512;
    float* kb   = p; p += (size_t)M * 128;
    float* vb   = p; p += (size_t)M * 128;
    float* ob   = p; p += (size_t)M * 512;
    float* ffn  = p; p += (size_t)M * 2048;   // also qkv buffer (ld 1536) on even layers
    float* xcat = p; p += (size_t)M * 1024;
    float* gb   = p; p += (size_t)M * 512;
    float* msc  = p; p += (size_t)4 * 20 * 512;
    float* ctx  = p; p += (size_t)4 * 20 * 512;
    float* cm   = p; p += (size_t)4 * 512;
    float* mi   = p; p += (size_t)4 * 512;
    float* feats= p; p += (size_t)4 * 32 * 64;
    float* mft  = p; p += (size_t)4 * 64;
    float* fa   = p; p += (size_t)4 * 512;
    float* fo   = p; p += (size_t)4 * 512;
    float* a1   = p; p += (size_t)4 * 256;
    float* a2   = p; p += (size_t)4 * 128;

    // ---- embedding + positional ----
    embed_k<<<2048, 128, 0, stream>>>(x, tok_emb, pe_w, pe_b, pos, z);

    // ---- memory-token attention + gated injection ----
    mem_score_k<<<80, 128, 0, stream>>>(memtok, z, msc);
    mem_ctx_k<<<80, 128, 0, stream>>>(z, msc, ctx);
    mem_mean_k<<<(2048 + 127) / 128, 128, 0, stream>>>(ctx, cm);
    launch_gemm(stream, cm, 512, mem_pw, 512, mi, 512, mem_pb, nullptr, 4, 512, 512, ACT_NONE);
    xcat_k<<<(int)(((size_t)M * 1024 + 255) / 256), 256, 0, stream>>>(z, mi, xcat);
    launch_gemm(stream, xcat, 1024, gate_w, 512, gb, 512, gate_b, nullptr, M, 512, 1024, ACT_SIGM);
    gateupd_k<<<(int)(((size_t)M * 512 + 255) / 256), 256, 0, stream>>>(z, gb, mi);

    // ---- transformer layers ----
    int fi = 0;
    for (int i = 0; i < 8; ++i) {
        int j = i / 2;
        if (i % 2 == 0) {
            // full-attention MHA block
            layernorm_k<<<2048, 128, 0, stream>>>(z, h, hn_w + j * 512, hn_b + j * 512, 512);
            launch_gemm(stream, h, 512, h_inw + (size_t)j * 512 * 1536, 1536,
                        ffn, 1536, h_inb + j * 1536, nullptr, M, 1536, 512, ACT_NONE);
            attn_wmma<<<dim3(64, 32), 128, 0, stream>>>(
                ffn, 1536, ffn + 512, 1536, ffn + 1024, 1536, ob,
                /*kvdiv=*/1, /*causal=*/0, nullptr, 0.17677669529f /* 1/sqrt(32) */);
            launch_gemm(stream, ob, 512, h_outw + (size_t)j * 512 * 512, 512,
                        z, 512, h_outb + j * 512, z, M, 512, 512, ACT_NONE);
        } else {
            // causal GQA + FFN block
            layernorm_k<<<2048, 128, 0, stream>>>(z, h, g_n1w + j * 512, g_n1b + j * 512, 512);
            launch_gemm(stream, h, 512, g_qw + (size_t)j * 512 * 512, 512,
                        qb, 512, nullptr, nullptr, M, 512, 512, ACT_NONE);
            launch_gemm(stream, h, 512, g_kw + (size_t)j * 512 * 128, 128,
                        kb, 128, nullptr, nullptr, M, 128, 512, ACT_NONE);
            launch_gemm(stream, h, 512, g_vw + (size_t)j * 512 * 128, 128,
                        vb, 128, nullptr, nullptr, M, 128, 512, ACT_NONE);
            qknorm_k<<<(2048 * 16 + 127) / 128, 128, 0, stream>>>(qb, g_qnw + j * 32, M, 16, 512);
            qknorm_k<<<(2048 * 4 + 127) / 128, 128, 0, stream>>>(kb, g_knw + j * 32, M, 4, 128);
            attn_wmma<<<dim3(64, 32), 128, 0, stream>>>(
                qb, 512, kb, 128, vb, 128, ob,
                /*kvdiv=*/4, /*causal=*/1, g_temp + j, 0.0f);
            launch_gemm(stream, ob, 512, g_ow + (size_t)j * 512 * 512, 512,
                        z, 512, nullptr, z, M, 512, 512, ACT_NONE);
            layernorm_k<<<2048, 128, 0, stream>>>(z, h, g_n2w + j * 512, g_n2b + j * 512, 512);
            layernorm_k<<<2048, 128, 0, stream>>>(h, h, g_lnw + j * 512, g_lnb + j * 512, 512);
            launch_gemm(stream, h, 512, g_w1 + (size_t)j * 512 * 2048, 2048,
                        ffn, 2048, g_b1 + j * 2048, nullptr, M, 2048, 512, ACT_SILU);
            launch_gemm(stream, ffn, 2048, g_w2 + (size_t)j * 2048 * 512, 512,
                        z, 512, g_b2 + j * 512, z, M, 512, 2048, ACT_NONE);
        }
        if ((i + 1) % 3 == 0 && fi < 2) {
            dft_k<<<(4 * 32 * 64 + 127) / 128, 128, 0, stream>>>(z, feats);
            fmean_k<<<2, 128, 0, stream>>>(feats, mft);
            launch_gemm(stream, mft, 64, f_fw + (size_t)fi * 64 * 512, 512,
                        fa, 512, f_fb + fi * 512, nullptr, 4, 512, 64, ACT_NONE);
            launch_gemm(stream, fa, 512, f_ow + (size_t)fi * 512 * 512, 512,
                        fo, 512, f_ob + fi * 512, nullptr, 4, 512, 512, ACT_NONE);
            addbc_k<<<(int)(((size_t)M * 512 + 255) / 256), 256, 0, stream>>>(z, fo);
            fi++;
        }
    }

    // ---- final LN directly into the d_out z-region ----
    float* preds = (float*)d_out;                   // (4 heads, 4 batch, 1024)
    float* zout  = (float*)d_out + 4 * 4 * 1024;    // (4, 512, 512)
    layernorm_k<<<2048, 128, 0, stream>>>(z, zout, lnf_w, lnf_b, 512);

    // ---- per-head MLP heads on last token; row stride T*D selects z[:, -1, :] ----
    const float* last = zout + (size_t)511 * 512;
    for (int hh = 0; hh < 4; ++hh) {
        launch_gemm(stream, last, 512 * 512, hd_w1 + (size_t)hh * 512 * 256, 256,
                    a1, 256, hd_b1 + hh * 256, nullptr, 4, 256, 512, ACT_SILU);
        launch_gemm(stream, a1, 256, hd_w2 + (size_t)hh * 256 * 128, 128,
                    a2, 128, hd_b2 + hh * 128, nullptr, 4, 128, 256, ACT_SILU);
        launch_gemm(stream, a2, 128, hd_w3 + (size_t)hh * 128 * 1024, 1024,
                    preds + (size_t)hh * 4 * 1024, 1024, hd_b3 + hh * 1024, nullptr,
                    4, 1024, 128, ACT_NONE);
    }
}
